// _HFNaiveMoE_62895501082712
// MI455X (gfx1250) — matrix operand order repaired and checked by reading, not compiled, not run on hardware
//
#include <hip/hip_runtime.h>
#include <hip/hip_bf16.h>

// Problem constants (from reference setup_inputs)
constexpr int T_ = 2048;   // tokens
constexpr int K_ = 6;      // top-k
constexpr int E_ = 32;     // experts
constexpr int H_ = 2048;   // hidden
constexpr int I_ = 1408;   // intermediate

typedef __attribute__((ext_vector_type(16))) __bf16 bf16x16;
typedef __attribute__((ext_vector_type(8)))  float  f32x8;

// ---------------- fragment loaders (wave32 WMMA layouts, CDNA5 ISA) ---

// A fragment, 16-bit 16x32 (MxK): lane m = lane&15; kbase = (lane<16)?0:8.
// Lane needs K = kbase+0..7 and kbase+16..23 -> two contiguous 16B chunks.
__device__ inline bf16x16 load_a_bf16(const __bf16* __restrict__ row, int kofs) {
  union { bf16x16 v; uint4 q[2]; } u;
  u.q[0] = *(const uint4*)(row + kofs);
  u.q[1] = *(const uint4*)(row + kofs + 16);
  return u.v;
}

// B fragment, 16-bit 32x16 (KxN): lane n = lane&15; kb = (lane<16)?0:8.
// Lane needs K = kb+0..7 and kb+16..23 (same as A); source is f32, convert to bf16 on the fly
// (pairs lower to v_cvt_pk_bf16_f32).
__device__ inline bf16x16 load_b_f32(const float* __restrict__ col) {
  bf16x16 b;
#pragma unroll
  for (int i = 0; i < 16; ++i) b[i] = (__bf16)col[i < 8 ? i : i + 8];
  return b;
}

__device__ inline f32x8 wmma_bf16(bf16x16 a, bf16x16 b, f32x8 c) {
  return __builtin_amdgcn_wmma_f32_16x16x32_bf16(false, a, false, b,
                                                 (short)0, c, false, false);
}

// fast silu: v_exp_f32 + v_rcp_f32 (avoid IEEE div sequence)
__device__ inline float fast_silu(float g) {
  return g * __builtin_amdgcn_rcpf(1.0f + __expf(-g));
}

// ---------------- routing kernels -------------------------------------------------

__global__ void moe_zero_init(const float* __restrict__ x, float* __restrict__ out,
                              __bf16* __restrict__ xbf,
                              int* __restrict__ counts, int* __restrict__ cursor) {
  int idx = blockIdx.x * blockDim.x + threadIdx.x;
  if (idx < T_ * H_) {
    out[idx] = 0.0f;
    xbf[idx] = (__bf16)x[idx];
  }
  if (idx < E_) { counts[idx] = 0; cursor[idx] = 0; }
}

__global__ void moe_count(const long long* __restrict__ ti, int* __restrict__ counts) {
  int i = blockIdx.x * blockDim.x + threadIdx.x;
  if (i < T_ * K_) atomicAdd(&counts[(int)ti[i]], 1);
}

__global__ void moe_scan(const int* __restrict__ counts, int* __restrict__ bases) {
  if (threadIdx.x == 0 && blockIdx.x == 0) {
    int s = 0;
    for (int e = 0; e < E_; ++e) { bases[e] = s; s += counts[e]; }
    bases[E_] = s;
  }
}

__global__ void moe_scatter(const long long* __restrict__ ti, const float* __restrict__ tw,
                            const int* __restrict__ bases, int* __restrict__ cursor,
                            int* __restrict__ token_ids, float* __restrict__ wts) {
  int i = blockIdx.x * blockDim.x + threadIdx.x;
  if (i < T_ * K_) {
    int e = (int)ti[i];
    int slot = atomicAdd(&cursor[e], 1);
    int pos = bases[e] + slot;
    token_ids[pos] = i / K_;
    wts[pos] = tw[i];
  }
}

// ---------------- GEMM 1: act = silu(x@Wg^T) * (x@Wu^T) * route_weight ------------
// grid = (I/128, E), block = 256 (8 waves). Each wave owns a 32(M) x 16(N) tile,
// computing gate and up together (A fragments reused across gate/up B fragments,
// B fragments reused across the two M sub-tiles): 4 wmma per 12 b128 loads.
__global__ __launch_bounds__(256)
void moe_gateup_silu(const __bf16* __restrict__ xbf,
                     const float* __restrict__ gup,
                     const int* __restrict__ counts,
                     const int* __restrict__ bases,
                     const int* __restrict__ token_ids,
                     const float* __restrict__ wts,
                     __bf16* __restrict__ act) {
  const int e = blockIdx.y;
  const int cnt = counts[e];
  if (cnt == 0) return;
  const int rbase = bases[e];

  const int lane  = threadIdx.x & 31;
  const int wave  = threadIdx.x >> 5;              // 0..7
  const int n0    = blockIdx.x * 128 + wave * 16;  // column into I
  const int nlane = lane & 15;

  const float* Wg = gup + (size_t)e * (2 * I_) * H_ + (size_t)(n0 + nlane) * H_;
  const float* Wu = Wg + (size_t)I_ * H_;
  const int akb = (lane < 16) ? 0 : 8;
  const int bkb = (lane < 16) ? 0 : 8;

  for (int m0 = 0; m0 < cnt; m0 += 32) {
    const int ra = m0 + (lane & 15);
    const int rb = ra + 16;
    const __bf16* A0 = xbf + (size_t)token_ids[rbase + (ra < cnt ? ra : cnt - 1)] * H_;
    const __bf16* A1 = xbf + (size_t)token_ids[rbase + (rb < cnt ? rb : cnt - 1)] * H_;

    f32x8 accg0 = {}, accg1 = {}, accu0 = {}, accu1 = {};
    for (int k0 = 0; k0 < H_; k0 += 32) {
      __builtin_prefetch(Wg + k0 + 256, 0, 3);     // global_prefetch_b8, ~8 steps ahead
      __builtin_prefetch(Wu + k0 + 256, 0, 3);
      bf16x16 a0 = load_a_bf16(A0, k0 + akb);
      bf16x16 a1 = load_a_bf16(A1, k0 + akb);
      bf16x16 bg = load_b_f32(Wg + k0 + bkb);
      bf16x16 bu = load_b_f32(Wu + k0 + bkb);
      accg0 = wmma_bf16(a0, bg, accg0);
      accu0 = wmma_bf16(a0, bu, accu0);
      accg1 = wmma_bf16(a1, bg, accg1);
      accu1 = wmma_bf16(a1, bu, accu1);
    }

    // C/D layout: VGPR r -> row r (lanes 0-15) / r+8 (lanes 16-31); lane&15 -> col.
    const int col  = n0 + nlane;
    const int rofs = (lane < 16) ? 0 : 8;
    const bool full = (m0 + 32 <= cnt);
#pragma unroll
    for (int half = 0; half < 2; ++half) {
      f32x8 ag = half ? accg1 : accg0;
      f32x8 au = half ? accu1 : accu0;
#pragma unroll
      for (int r = 0; r < 8; ++r) {
        int m = m0 + half * 16 + r + rofs;
        if (full || m < cnt) {
          float v = fast_silu(ag[r]) * au[r] * wts[rbase + m];
          act[(size_t)(rbase + m) * I_ + col] = (__bf16)v;
        }
      }
    }
  }
}

// ---------------- GEMM 2: out[token] += act @ down^T ------------------------------
// grid = (H/256, E), block = 256 (8 waves). Each wave owns a 32(M) x 32(N) tile:
// 4 wmma per 12 b128 loads. Scatter-add into out with f32 atomics.
__global__ __launch_bounds__(256)
void moe_down_scatter(const __bf16* __restrict__ act,
                      const float* __restrict__ down,
                      const int* __restrict__ counts,
                      const int* __restrict__ bases,
                      const int* __restrict__ token_ids,
                      float* __restrict__ out) {
  const int e = blockIdx.y;
  const int cnt = counts[e];
  if (cnt == 0) return;
  const int rbase = bases[e];

  const int lane  = threadIdx.x & 31;
  const int wave  = threadIdx.x >> 5;
  const int n0    = blockIdx.x * 256 + wave * 32;  // column into H
  const int nlane = lane & 15;

  const float* W0 = down + (size_t)e * H_ * I_ + (size_t)(n0 + nlane) * I_;
  const float* W1 = W0 + (size_t)16 * I_;
  const int akb = (lane < 16) ? 0 : 8;
  const int bkb = (lane < 16) ? 0 : 8;

  for (int m0 = 0; m0 < cnt; m0 += 32) {
    const int ra = m0 + (lane & 15);
    const int rb = ra + 16;
    const __bf16* A0 = act + (size_t)(rbase + (ra < cnt ? ra : cnt - 1)) * I_;
    const __bf16* A1 = act + (size_t)(rbase + (rb < cnt ? rb : cnt - 1)) * I_;

    f32x8 acc00 = {}, acc01 = {}, acc10 = {}, acc11 = {};
    for (int k0 = 0; k0 < I_; k0 += 32) {
      __builtin_prefetch(W0 + k0 + 256, 0, 3);
      __builtin_prefetch(W1 + k0 + 256, 0, 3);
      bf16x16 a0 = load_a_bf16(A0, k0 + akb);
      bf16x16 a1 = load_a_bf16(A1, k0 + akb);
      bf16x16 b0 = load_b_f32(W0 + k0 + bkb);
      bf16x16 b1 = load_b_f32(W1 + k0 + bkb);
      acc00 = wmma_bf16(a0, b0, acc00);
      acc01 = wmma_bf16(a0, b1, acc01);
      acc10 = wmma_bf16(a1, b0, acc10);
      acc11 = wmma_bf16(a1, b1, acc11);
    }

    const int rofs = (lane < 16) ? 0 : 8;
    const bool full = (m0 + 32 <= cnt);
#pragma unroll
    for (int half = 0; half < 2; ++half) {
      f32x8 c0 = half ? acc10 : acc00;
      f32x8 c1 = half ? acc11 : acc01;
#pragma unroll
      for (int r = 0; r < 8; ++r) {
        int m = m0 + half * 16 + r + rofs;
        if (full || m < cnt) {
          size_t trow = (size_t)token_ids[rbase + m] * H_;
          atomicAdd(&out[trow + n0 + nlane],      c0[r]);
          atomicAdd(&out[trow + n0 + 16 + nlane], c1[r]);
        }
      }
    }
  }
}

// ---------------- host launcher ---------------------------------------------------

extern "C" void kernel_launch(void* const* d_in, const int* in_sizes, int n_in,
                              void* d_out, int out_size, void* d_ws, size_t ws_size,
                              hipStream_t stream) {
  const float*     x   = (const float*)d_in[0];       // (T,H) f32
  const long long* ti  = (const long long*)d_in[1];   // (T,K) i64
  const float*     tw  = (const float*)d_in[2];       // (T,K) f32
  const float*     gup = (const float*)d_in[3];       // (E,2I,H) f32
  const float*     dwn = (const float*)d_in[4];       // (E,H,I) f32
  float* out = (float*)d_out;                         // (T,H) f32

  // workspace carving (all 16B aligned)
  char* w = (char*)d_ws;
  int*    counts    = (int*)(w + 0);                          // E
  int*    cursor    = (int*)(w + 128);                        // E
  int*    bases     = (int*)(w + 256);                        // E+1
  int*    token_ids = (int*)(w + 512);                        // T*K
  float*  wts       = (float*)(w + 512 + 4 * T_ * K_);        // T*K
  __bf16* xbf       = (__bf16*)(w + 512 + 8 * T_ * K_);       // T*H bf16
  __bf16* act       = xbf + (size_t)T_ * H_;                  // T*K*I bf16

  // 1) zero out + routing counters, convert hidden_states to bf16
  {
    int n = T_ * H_;
    moe_zero_init<<<(n + 255) / 256, 256, 0, stream>>>(x, out, xbf, counts, cursor);
  }
  // 2-4) route tokens to experts (count -> scan -> compacted scatter)
  {
    int n = T_ * K_;
    moe_count<<<(n + 255) / 256, 256, 0, stream>>>(ti, counts);
    moe_scan<<<1, 32, 0, stream>>>(counts, bases);
    moe_scatter<<<(n + 255) / 256, 256, 0, stream>>>(ti, tw, bases, cursor,
                                                     token_ids, wts);
  }
  // 5) gate_up GEMM + SiLU + routing weight -> act (bf16)
  {
    dim3 grid(I_ / 128, E_);
    moe_gateup_silu<<<grid, 256, 0, stream>>>(xbf, gup, counts, bases,
                                              token_ids, wts, act);
  }
  // 6) down GEMM, scatter-add into out
  {
    dim3 grid(H_ / 256, E_);
    moe_down_scatter<<<grid, 256, 0, stream>>>(act, dwn, counts, bases,
                                               token_ids, out);
  }
}
